// Attention_Branch_50483045597357
// MI455X (gfx1250) — matrix operand order repaired and checked by reading, not compiled, not run on hardware
//
#include <hip/hip_runtime.h>

typedef __attribute__((ext_vector_type(16))) __bf16 v16bf;
typedef __attribute__((ext_vector_type(8)))  float  v8f;

__device__ __forceinline__ __bf16 f2bf(float f) {
  union { float f; unsigned u; } a; a.f = f;
  unsigned r = a.u + 0x7FFFu + ((a.u >> 16) & 1u);   // round-to-nearest-even
  unsigned short h = (unsigned short)(r >> 16);
  __bf16 o; __builtin_memcpy(&o, &h, 2); return o;
}
__device__ __forceinline__ float lrelu_f(float v){ return v > 0.f ? v : 0.2f*v; }

// ---------------------------------------------------------------------------
// Pack OIHW f32 weights into WMMA 16-bit A-matrix (16x32) fragments, in contiguous k order (same as B).
// Fragment storage: [((tap*NCB+cb)*MT+m)*32 + lane]*16 + i  (bf16 elements)
// lane L: M = L%16 ; element i: K = (L>=16)*16 + i
// ---------------------------------------------------------------------------
__global__ void pack_wmma_a(const float* __restrict__ w, __bf16* __restrict__ out,
                            int Cout, int Cin, int KHW, int MT, int NCB, int total)
{
  int idx = blockIdx.x*blockDim.x + threadIdx.x;
  if (idx >= total) return;
  int i    = idx & 15;
  int lane = (idx >> 4) & 31;
  int rest = idx >> 9;
  int m    = rest % MT;
  int qq   = rest / MT;
  int cb   = qq % NCB;
  int tap  = qq / NCB;
  int M    = lane & 15;
  int base = (lane >> 4) * 16;
  int K    = base + i;
  int co = m*16 + M;
  int ci = cb*32 + K;
  float v = 0.f;
  if (co < Cout && ci < Cin)
    v = w[(long)(co*Cin + ci)*KHW + tap];
  out[idx] = f2bf(v);
}

// ---------------------------------------------------------------------------
// Generic implicit-GEMM conv, NHWC bf16 in -> NCHW f32 out, via WMMA bf16.
// One wave handles a 16-pixel strip of one output row, all Cout channels.
// MT = Cout/16 tiles (>=1), NCB = Cin/32 blocks.
// ---------------------------------------------------------------------------
template<int MT, int NCB>
__global__ __launch_bounds__(256) void conv_wmma(
    const __bf16* __restrict__ in,   // B x Hi x Wi x (32*NCB), NHWC bf16
    const __bf16* __restrict__ wpk,  // packed A fragments
    float* __restrict__ out,         // B x Cout x Ho x Wo, NCHW f32
    int B_, int Hi, int Wi, int Ho, int Wo,
    int KH, int KW, int stride, int pad, int Cout)
{
  const int Cin  = 32*NCB;
  const int lane = threadIdx.x & 31;
  const int wv   = threadIdx.x >> 5;
  const int sx_n = (Wo + 15) >> 4;
  long gs = (long)blockIdx.x * (blockDim.x >> 5) + wv;
  long ns = (long)B_ * Ho * sx_n;
  if (gs >= ns) return;                       // wave-uniform exit
  int  sx = (int)(gs % sx_n); long t_ = gs / sx_n;
  int  ho = (int)(t_ % Ho);   int  b  = (int)(t_ / Ho);
  const int  n    = lane & 15;
  const int  half = lane >> 4;
  const int  wo   = (sx << 4) + n;
  const bool wok  = wo < Wo;

  v8f acc[MT] = {};
  const __bf16* inb = in + (long)b * Hi * Wi * Cin;

  for (int ky = 0; ky < KH; ++ky) {
    int hi = ho*stride + ky - pad;
    if (hi < 0 || hi >= Hi) continue;         // wave-uniform
    const __bf16* rowp = inb + (long)hi * Wi * Cin;
    // prefetch next tap row for this strip
    if (ky + 1 < KH) {
      int hin = ho*stride + ky + 1 - pad;
      if (hin >= 0 && hin < Hi && wok) {
        int wic = wo*stride; if (wic >= Wi) wic = Wi - 1; if (wic < 0) wic = 0;
        __builtin_prefetch(inb + ((long)hin * Wi + wic) * Cin, 0, 1);
      }
    }
    for (int kx = 0; kx < KW; ++kx) {
      int  wi = wo*stride + kx - pad;
      bool ok = wok && (wi >= 0) && (wi < Wi);
      const __bf16* bp = rowp + (long)wi * Cin + half*16;
      const int tap = ky*KW + kx;
      #pragma unroll
      for (int cb = 0; cb < NCB; ++cb) {
        v16bf bv = {};
        if (ok) bv = *(const v16bf*)(bp + cb*32);   // 32B contiguous (2x b128)
        const __bf16* ap = wpk + (long)((tap*NCB + cb)*MT)*512 + lane*16;
        #pragma unroll
        for (int m = 0; m < MT; ++m) {
          v16bf av = *(const v16bf*)(ap + m*512);
          acc[m] = __builtin_amdgcn_wmma_f32_16x16x32_bf16(
                     false, av, false, bv, (short)0, acc[m], false, false);
        }
      }
    }
  }

  if (wok) {
    float* ob = out + (long)b*Cout*Ho*Wo + (long)ho*Wo + wo;
    #pragma unroll
    for (int m = 0; m < MT; ++m) {
      #pragma unroll
      for (int r = 0; r < 8; ++r) {
        int co = m*16 + half*8 + r;             // C/D layout: lanes16-31 -> M+8
        if (co < Cout) ob[(long)co*Ho*Wo] = acc[m][r];
      }
    }
  }
}

// ---------------------------------------------------------------------------
// q/k: 1x1 conv 32->8 + L2 normalize (per-pixel)
// ---------------------------------------------------------------------------
__global__ __launch_bounds__(256) void qk_kernel(
    const float* __restrict__ x, const float* __restrict__ wq, const float* __restrict__ wk,
    float* __restrict__ q, float* __restrict__ k, int B_, int HW)
{
  __shared__ float swq[256], swk[256];
  int tid = threadIdx.x;
  swq[tid] = wq[tid]; swk[tid] = wk[tid];
  __syncthreads();
  long p_ = (long)blockIdx.x*blockDim.x + tid;
  if (p_ >= (long)B_*HW) return;
  long b = p_ / HW, hw = p_ % HW;
  const float* xb = x + b*32*HW + hw;
  float xi[32];
  #pragma unroll
  for (int c = 0; c < 32; ++c) xi[c] = xb[(long)c*HW];
  float qa[8], ka[8], qn = 0.f, kn = 0.f;
  #pragma unroll
  for (int f = 0; f < 8; ++f) {
    float aq = 0.f, ak = 0.f;
    #pragma unroll
    for (int c = 0; c < 32; ++c) { aq += swq[f*32+c]*xi[c]; ak += swk[f*32+c]*xi[c]; }
    qa[f] = aq; ka[f] = ak; qn += aq*aq; kn += ak*ak;
  }
  float qi = 1.f/(sqrtf(qn)+1e-6f), ki = 1.f/(sqrtf(kn)+1e-6f);
  float* qb = q + b*8*HW + hw; float* kb = k + b*8*HW + hw;
  #pragma unroll
  for (int f = 0; f < 8; ++f) { qb[(long)f*HW] = qa[f]*qi; kb[(long)f*HW] = ka[f]*ki; }
}

// ---------------------------------------------------------------------------
// Fused shift-attention: 25-shift dots, smooth-normalize, grouped 3x3
// (w_direct), 1x1 (w_nl) + lrelu  ->  y (NCHW f32).  LDS tiled, 16x16 pixels.
// ---------------------------------------------------------------------------
__global__ __launch_bounds__(256) void attn_y_kernel(
    const float* __restrict__ q, const float* __restrict__ k,
    const float* __restrict__ wd, const float* __restrict__ wnl,
    float* __restrict__ y, int H, int W)
{
  __shared__ float qs[8][22][22];   // halo radius 3 (shift 2 + conv 1)
  __shared__ float ks8[8][18][18];  // halo radius 1
  __shared__ float wdS[25*81];
  __shared__ float wnlS[32*25];

  const int HW = H*W;
  const int tx_n = W >> 4, ty_n = H >> 4;
  int bid = blockIdx.x;
  int tilex = bid % tx_n; bid /= tx_n;
  int tiley = bid % ty_n; int b = bid / ty_n;
  int h0 = tiley*16, w0 = tilex*16;
  int tid = threadIdx.y*16 + threadIdx.x;
  const float* qb = q + (long)b*8*HW;
  const float* kb = k + (long)b*8*HW;

  for (int i = tid; i < 8*22*22; i += 256) {
    int c = i/484, r2 = i%484, yy = r2/22, xx = r2%22;
    int gh = h0 - 3 + yy, gw = w0 - 3 + xx;
    float v = 0.f;
    if (gh >= 0 && gh < H && gw >= 0 && gw < W) v = qb[(long)c*HW + gh*W + gw];
    qs[c][yy][xx] = v;
  }
  for (int i = tid; i < 8*18*18; i += 256) {
    int c = i/324, r2 = i%324, yy = r2/18, xx = r2%18;
    int gh = h0 - 1 + yy, gw = w0 - 1 + xx;
    float v = 0.f;
    if (gh >= 0 && gh < H && gw >= 0 && gw < W) v = kb[(long)c*HW + gh*W + gw];
    ks8[c][yy][xx] = v;
  }
  for (int i = tid; i < 25*81; i += 256) wdS[i] = wd[i];
  for (int i = tid; i < 32*25; i += 256) wnlS[i] = wnl[i];
  __syncthreads();

  int lx = threadIdx.x, ly = threadIdx.y;
  int h = h0 + ly, w = w0 + lx;

  float mixed[25];
  #pragma unroll
  for (int s = 0; s < 25; ++s) mixed[s] = 0.f;

  for (int dy = -1; dy <= 1; ++dy) {
    int hh = h + dy; if (hh < 0 || hh >= H) continue;
    for (int dx = -1; dx <= 1; ++dx) {
      int ww = w + dx; if (ww < 0 || ww >= W) continue;
      int kly = ly + 1 + dy, klx = lx + 1 + dx;
      int qly = ly + 3 + dy, qlx = lx + 3 + dx;
      float kv[8];
      #pragma unroll
      for (int c = 0; c < 8; ++c) kv[c] = ks8[c][kly][klx];
      float dots[25]; float dmax = -1e30f;
      #pragma unroll
      for (int s = 0; s < 25; ++s) {
        const int si = s/5 - 2, sj = s%5 - 2;
        float d = 0.f;
        #pragma unroll
        for (int c = 0; c < 8; ++c) d += qs[c][qly-si][qlx-sj] * kv[c];
        d = 1.f - d;               // dot = 1 - <shifted_q, k>
        dots[s] = d;
        dmax = fmaxf(dmax, d);
      }
      float inv = 1.f/(dmax + 1e-6f);
      int tap = (dy+1)*3 + (dx+1);
      #pragma unroll
      for (int s = 0; s < 25; ++s) {
        const int si = s/5 - 2, sj = s%5 - 2;
        const float* wp = &wdS[s*81 + tap];        // wdS[s*81 + c*9 + tap]
        float acc = wp[72] * (1.f - dots[s]*inv*4.f);  // /H_SMOOTH(0.25)
        #pragma unroll
        for (int c = 0; c < 8; ++c)
          acc += wp[c*9] * (qs[c][qly-si][qlx-sj] - kv[c]);
        mixed[s] += acc;
      }
    }
  }
  float* yb = y + (long)b*32*HW + (long)h*W + w;
  #pragma unroll
  for (int f = 0; f < 32; ++f) {
    float a = 0.f;
    #pragma unroll
    for (int s = 0; s < 25; ++s) a += wnlS[f*25+s]*mixed[s];
    yb[(long)f*HW] = lrelu_f(a);
  }
}

// ---------------------------------------------------------------------------
// Instance norm: per-(b,c) stats then normalize + lrelu
// ---------------------------------------------------------------------------
__global__ __launch_bounds__(256) void inorm_stats(
    const float* __restrict__ t, float* __restrict__ st, int HW)
{
  __shared__ float sA[256], sB[256];
  int bc = blockIdx.x;
  const float* p_ = t + (long)bc*HW;
  float s = 0.f, s2 = 0.f;
  for (int i = threadIdx.x; i < HW; i += 256) { float v = p_[i]; s += v; s2 += v*v; }
  sA[threadIdx.x] = s; sB[threadIdx.x] = s2;
  __syncthreads();
  for (int off = 128; off > 0; off >>= 1) {
    if (threadIdx.x < off) {
      sA[threadIdx.x] += sA[threadIdx.x+off];
      sB[threadIdx.x] += sB[threadIdx.x+off];
    }
    __syncthreads();
  }
  if (threadIdx.x == 0) {
    float m = sA[0]/HW;
    float v = sB[0]/HW - m*m;
    st[2*bc]   = m;
    st[2*bc+1] = rsqrtf(v + 1e-5f);
  }
}

__global__ void inorm_apply(const float* __restrict__ t, const float* __restrict__ st,
                            float* __restrict__ o, int HW, long total)
{
  long idx = (long)blockIdx.x*blockDim.x + threadIdx.x;
  if (idx >= total) return;
  long bc = idx / HW;
  float v = (t[idx] - st[2*bc]) * st[2*bc+1];
  o[idx] = lrelu_f(v);
}

// ---------------------------------------------------------------------------
// NCHW f32 -> NHWC bf16 conversions (with optional upsample / concat folded)
// ---------------------------------------------------------------------------
__global__ void to_nhwc_bf16(const float* __restrict__ src, __bf16* __restrict__ dst,
                             int C, int HW, long total)
{
  long idx = (long)blockIdx.x*blockDim.x + threadIdx.x;
  if (idx >= total) return;
  int c = (int)(idx % C);
  long bp = idx / C;
  long b = bp / HW, hw = bp % HW;
  dst[idx] = f2bf(src[(b*C + c)*HW + hw]);
}

__global__ void concat_to_nhwc_bf16(const float* __restrict__ s1, const float* __restrict__ s2,
                                    __bf16* __restrict__ dst, int HW, long total) // C = 64
{
  long idx = (long)blockIdx.x*blockDim.x + threadIdx.x;
  if (idx >= total) return;
  int c = (int)(idx & 63);
  long bp = idx >> 6;
  long b = bp / HW, hw = bp % HW;
  const float* s = (c < 32) ? s1 : s2;
  dst[idx] = f2bf(s[(b*32 + (c & 31))*HW + hw]);
}

__global__ void up2x_to_nhwc_bf16(const float* __restrict__ src, __bf16* __restrict__ dst,
                                  int C, int Ho, int Wo, long total) // src at Ho/2 x Wo/2
{
  long idx = (long)blockIdx.x*blockDim.x + threadIdx.x;
  if (idx >= total) return;
  int c = (int)(idx % C);
  long bp = idx / C;
  int w = (int)(bp % Wo); long t = bp / Wo; int h = (int)(t % Ho); long b = t / Ho;
  int Hs = Ho >> 1, Ws = Wo >> 1;
  dst[idx] = f2bf(src[((b*C + c)*Hs + (h>>1))*Ws + (w>>1)]);
}

__global__ void up2x_concat_to_nhwc_bf16(const float* __restrict__ s1, const float* __restrict__ s2,
                                         __bf16* __restrict__ dst, int Ho, int Wo, long total) // C=64
{
  long idx = (long)blockIdx.x*blockDim.x + threadIdx.x;
  if (idx >= total) return;
  int c = (int)(idx & 63);
  long bp = idx >> 6;
  int w = (int)(bp % Wo); long t = bp / Wo; int h = (int)(t % Ho); long b = t / Ho;
  int Hs = Ho >> 1, Ws = Wo >> 1;
  const float* s = (c < 32) ? s1 : s2;
  dst[idx] = f2bf(s[((b*32 + (c & 31))*Hs + (h>>1))*Ws + (w>>1)]);
}

// ---------------------------------------------------------------------------
extern "C" void kernel_launch(void* const* d_in, const int* in_sizes, int n_in,
                              void* d_out, int out_size, void* d_ws, size_t ws_size,
                              hipStream_t stream)
{
  (void)in_sizes; (void)n_in; (void)out_size; (void)ws_size;
  const float* x        = (const float*)d_in[0];
  const float* w_q      = (const float*)d_in[1];
  const float* w_k      = (const float*)d_in[2];
  const float* w_direct = (const float*)d_in[3];
  const float* w_nl     = (const float*)d_in[4];
  const float* w_head   = (const float*)d_in[5];
  const float* w_down1  = (const float*)d_in[6];
  const float* w_down2  = (const float*)d_in[7];
  const float* w_e0     = (const float*)d_in[8];
  const float* w_up2    = (const float*)d_in[9];
  const float* w_up1    = (const float*)d_in[10];
  const float* w_tail   = (const float*)d_in[11];

  const int B = 4, F = 32, H = 320, W = 320;
  const int HW = H*W, H2 = 160, W2 = 160, HW2 = H2*W2, H4 = 80, W4 = 80, HW4 = H4*W4;

  char* p = (char*)d_ws;
  auto carve = [&](size_t bytes) -> void* {
    void* r = (void*)p; p += (bytes + 255) & ~(size_t)255; return r;
  };
  float* q   = (float*)carve((size_t)B*8*HW*4);
  float* k   = (float*)carve((size_t)B*8*HW*4);
  float* y   = (float*)carve((size_t)B*F*HW*4);
  float* t   = (float*)carve((size_t)B*F*HW*4);    // pre-norm temp (reused)
  float* d0  = (float*)carve((size_t)B*F*HW*4);
  float* d1  = (float*)carve((size_t)B*F*HW2*4);
  float* d2  = (float*)carve((size_t)B*F*HW4*4);
  float* e0  = (float*)carve((size_t)B*F*HW4*4);
  float* u2  = (float*)carve((size_t)B*F*HW2*4);
  float* u1  = (float*)carve((size_t)B*F*HW*4);
  float* st  = (float*)carve((size_t)2*B*F*4);
  __bf16* abf = (__bf16*)carve((size_t)B*HW*64*2); // NHWC bf16 staging (reused)
  __bf16* pw_head = (__bf16*)carve((size_t)9216*2);
  __bf16* pw_d1   = (__bf16*)carve((size_t)16384*2);
  __bf16* pw_d2   = (__bf16*)carve((size_t)16384*2);
  __bf16* pw_e0   = (__bf16*)carve((size_t)9216*2);
  __bf16* pw_u2   = (__bf16*)carve((size_t)9216*2);
  __bf16* pw_u1   = (__bf16*)carve((size_t)18432*2);
  __bf16* pw_tl   = (__bf16*)carve((size_t)9216*2);

  // --- weight packing into WMMA A layout ---
  pack_wmma_a<<<(9216 +255)/256,256,0,stream>>>(w_head,  pw_head, 32,32, 9, 2,1, 9216);
  pack_wmma_a<<<(16384+255)/256,256,0,stream>>>(w_down1, pw_d1,   32,32,16, 2,1,16384);
  pack_wmma_a<<<(16384+255)/256,256,0,stream>>>(w_down2, pw_d2,   32,32,16, 2,1,16384);
  pack_wmma_a<<<(9216 +255)/256,256,0,stream>>>(w_e0,    pw_e0,   32,32, 9, 2,1, 9216);
  pack_wmma_a<<<(9216 +255)/256,256,0,stream>>>(w_up2,   pw_u2,   32,32, 9, 2,1, 9216);
  pack_wmma_a<<<(18432+255)/256,256,0,stream>>>(w_up1,   pw_u1,   32,64, 9, 2,2,18432);
  pack_wmma_a<<<(9216 +255)/256,256,0,stream>>>(w_tail,  pw_tl,    1,64, 9, 1,2, 9216);

  // --- q/k + shift attention -> y ---
  qk_kernel<<<(B*HW+255)/256,256,0,stream>>>(x, w_q, w_k, q, k, B, HW);
  attn_y_kernel<<<B*(H/16)*(W/16), dim3(16,16), 0, stream>>>(q, k, w_direct, w_nl, y, H, W);

  // --- head: 3x3 s1 p1, 32->32 @320 ---
  {
    long tot = (long)B*HW*32;
    to_nhwc_bf16<<<(int)((tot+255)/256),256,0,stream>>>(y, abf, 32, HW, tot);
    int strips = B*H*((W+15)/16);
    conv_wmma<2,1><<<(strips+7)/8,256,0,stream>>>(abf, pw_head, t, B, H,W,H,W, 3,3,1,1, 32);
    inorm_stats<<<B*32,256,0,stream>>>(t, st, HW);
    long tt = (long)B*32*HW;
    inorm_apply<<<(int)((tt+255)/256),256,0,stream>>>(t, st, d0, HW, tt);
  }
  // --- down1: 4x4 s2 p1, 32->32, 320->160 ---
  {
    long tot = (long)B*HW*32;
    to_nhwc_bf16<<<(int)((tot+255)/256),256,0,stream>>>(d0, abf, 32, HW, tot);
    int strips = B*H2*((W2+15)/16);
    conv_wmma<2,1><<<(strips+7)/8,256,0,stream>>>(abf, pw_d1, t, B, H,W,H2,W2, 4,4,2,1, 32);
    inorm_stats<<<B*32,256,0,stream>>>(t, st, HW2);
    long tt = (long)B*32*HW2;
    inorm_apply<<<(int)((tt+255)/256),256,0,stream>>>(t, st, d1, HW2, tt);
  }
  // --- down2: 4x4 s2 p1, 32->32, 160->80 ---
  {
    long tot = (long)B*HW2*32;
    to_nhwc_bf16<<<(int)((tot+255)/256),256,0,stream>>>(d1, abf, 32, HW2, tot);
    int strips = B*H4*((W4+15)/16);
    conv_wmma<2,1><<<(strips+7)/8,256,0,stream>>>(abf, pw_d2, t, B, H2,W2,H4,W4, 4,4,2,1, 32);
    inorm_stats<<<B*32,256,0,stream>>>(t, st, HW4);
    long tt = (long)B*32*HW4;
    inorm_apply<<<(int)((tt+255)/256),256,0,stream>>>(t, st, d2, HW4, tt);
  }
  // --- e0: 3x3 s1 p1, 32->32 @80 ---
  {
    long tot = (long)B*HW4*32;
    to_nhwc_bf16<<<(int)((tot+255)/256),256,0,stream>>>(d2, abf, 32, HW4, tot);
    int strips = B*H4*((W4+15)/16);
    conv_wmma<2,1><<<(strips+7)/8,256,0,stream>>>(abf, pw_e0, t, B, H4,W4,H4,W4, 3,3,1,1, 32);
    inorm_stats<<<B*32,256,0,stream>>>(t, st, HW4);
    long tt = (long)B*32*HW4;
    inorm_apply<<<(int)((tt+255)/256),256,0,stream>>>(t, st, e0, HW4, tt);
  }
  // --- u2: up2x(e0) then 3x3 s1 p1 @160 ---
  {
    long tot = (long)B*HW2*32;
    up2x_to_nhwc_bf16<<<(int)((tot+255)/256),256,0,stream>>>(e0, abf, 32, H2, W2, tot);
    int strips = B*H2*((W2+15)/16);
    conv_wmma<2,1><<<(strips+7)/8,256,0,stream>>>(abf, pw_u2, t, B, H2,W2,H2,W2, 3,3,1,1, 32);
    inorm_stats<<<B*32,256,0,stream>>>(t, st, HW2);
    long tt = (long)B*32*HW2;
    inorm_apply<<<(int)((tt+255)/256),256,0,stream>>>(t, st, u2, HW2, tt);
  }
  // --- u1: up2x(concat(d1,u2)) then 3x3 s1 p1, 64->32 @320 ---
  {
    long tot = (long)B*HW*64;
    up2x_concat_to_nhwc_bf16<<<(int)((tot+255)/256),256,0,stream>>>(d1, u2, abf, H, W, tot);
    int strips = B*H*((W+15)/16);
    conv_wmma<2,2><<<(strips+7)/8,256,0,stream>>>(abf, pw_u1, t, B, H,W,H,W, 3,3,1,1, 32);
    inorm_stats<<<B*32,256,0,stream>>>(t, st, HW);
    long tt = (long)B*32*HW;
    inorm_apply<<<(int)((tt+255)/256),256,0,stream>>>(t, st, u1, HW, tt);
  }
  // --- tail: concat(d0,u1) then 3x3 s1 p1, 64->1 @320 -> d_out ---
  {
    long tot = (long)B*HW*64;
    concat_to_nhwc_bf16<<<(int)((tot+255)/256),256,0,stream>>>(d0, u1, abf, HW, tot);
    int strips = B*H*((W+15)/16);
    conv_wmma<1,2><<<(strips+7)/8,256,0,stream>>>(abf, pw_tl, (float*)d_out,
                                                  B, H,W,H,W, 3,3,1,1, 1);
  }
}
